// MultiHeadAttention_11922829213699
// MI455X (gfx1250) — matrix operand order repaired and checked by reading, not compiled, not run on hardware
//
#include <hip/hip_runtime.h>

// ---------------------------------------------------------------------------
// MI455X (gfx1250) multi-head attention.
// All matmuls run on v_wmma_f32_16x16x32_bf16 (bf16 A/B, f32 accumulate).
// Flash-attention inner loop never materializes the 2048x2048 score matrix.
// V is stored transposed [B,H,DK,T] so every WMMA fragment load in the
// attention hot loop is two contiguous 8-half (b128) runs per lane.
// Workspace requirement: ~92 MB (xb + wqkvb + woutb + Q + K + Vt + ctx, bf16).
// ---------------------------------------------------------------------------

typedef __attribute__((ext_vector_type(16))) __bf16 v16bf;
typedef __attribute__((ext_vector_type(8)))  float  v8f;

#define B_      4
#define T_      2048
#define DIM_    1024
#define HEADS_  16
#define DK_     64
#define MROWS_  (B_ * T_)        // 8192
#define NQKV_   (3 * DIM_)       // 3072

// fp32 -> bf16 round-to-nearest-even
static __device__ __forceinline__ unsigned short f2bf(float f) {
    unsigned int u = __float_as_uint(f);
    u += 0x7fffu + ((u >> 16) & 1u);
    return (unsigned short)(u >> 16);
}

static __device__ __forceinline__ v8f wmma_bf16(v16bf a, v16bf b, v8f c) {
    return __builtin_amdgcn_wmma_f32_16x16x32_bf16(
        /*neg_a=*/false, a, /*neg_b=*/false, b,
        /*c_mod=*/(short)0, c, /*reuse_a=*/false, /*reuse_b=*/false);
}

// A fragment: 16x32 (MxK), src row-major with leading dim ld.
// Lane L holds row m = L&15; half j -> k = ((j>>3)*16) + (L>>4)*8 + (j&7).
static __device__ __forceinline__ v16bf load_A(const unsigned short* src, int ld, int lane) {
    union { v16bf v; unsigned short u[16]; } r;
    const int m  = lane & 15;
    const int kg = (lane >> 4) * 8;
    const unsigned short* p = src + (size_t)m * ld;
#pragma unroll
    for (int j = 0; j < 16; ++j) {
        int k = ((j >> 3) * 16) + kg + (j & 7);
        r.u[j] = p[k];
    }
    return r.v;
}

// B fragment 32x16 (KxN) with B(k,n) = src[n*ld + k]  (src is N-major: computes X @ src^T)
// Lane L holds col n = L&15; half j -> k = ((j>>3)*16) + (L>>4)*8 + (j&7) (two 8-half runs).
static __device__ __forceinline__ v16bf load_BT(const unsigned short* src, int ld, int lane) {
    union { v16bf v; unsigned short u[16]; } r;
    const int n  = lane & 15;
    const int kg = (lane >> 4) * 8;
    const unsigned short* p = src + (size_t)n * ld;
#pragma unroll
    for (int j = 0; j < 16; ++j) r.u[j] = p[((j >> 3) * 16) + kg + (j & 7)];
    return r.v;
}

// ---------------------------------------------------------------------------
// fp32 -> bf16 cast (grid-stride)
// ---------------------------------------------------------------------------
__global__ void k_cast(const float* __restrict__ s, unsigned short* __restrict__ d, int n) {
    int i = blockIdx.x * blockDim.x + threadIdx.x;
    int stride = gridDim.x * blockDim.x;
    for (; i < n; i += stride) d[i] = f2bf(s[i]);
}

// ---------------------------------------------------------------------------
// QKV projection: qkv = x @ w_qkv^T + b_qkv, scattered into
//   Q,K : [B,H,T,64] bf16     V : transposed [B,H,64,T] bf16
// Block = 256 threads (8 waves), block tile 128(M) x 64(N), wave tile 32x32.
// ---------------------------------------------------------------------------
__global__ __launch_bounds__(256) void k_qkv(const unsigned short* __restrict__ xb,
                                             const unsigned short* __restrict__ wb,
                                             const float* __restrict__ bias,
                                             unsigned short* __restrict__ Q,
                                             unsigned short* __restrict__ K,
                                             unsigned short* __restrict__ Vt) {
    const int lane = threadIdx.x & 31;
    const int wid  = threadIdx.x >> 5;
    const int m0 = blockIdx.y * 128 + (wid & 3) * 32;
    const int n0 = blockIdx.x * 64  + (wid >> 2) * 32;

    v8f acc[2][2] = {};
    for (int k0 = 0; k0 < DIM_; k0 += 32) {
        v16bf a0 = load_A (xb + (size_t)m0        * DIM_ + k0, DIM_, lane);
        v16bf a1 = load_A (xb + (size_t)(m0 + 16) * DIM_ + k0, DIM_, lane);
        v16bf b0 = load_BT(wb + (size_t)n0        * DIM_ + k0, DIM_, lane);
        v16bf b1 = load_BT(wb + (size_t)(n0 + 16) * DIM_ + k0, DIM_, lane);
        acc[0][0] = wmma_bf16(a0, b0, acc[0][0]);
        acc[0][1] = wmma_bf16(a0, b1, acc[0][1]);
        acc[1][0] = wmma_bf16(a1, b0, acc[1][0]);
        acc[1][1] = wmma_bf16(a1, b1, acc[1][1]);
    }

    const int ncol = lane & 15;
    const int mb   = (lane >> 4) * 8;
#pragma unroll
    for (int i = 0; i < 2; ++i) {
#pragma unroll
        for (int j = 0; j < 2; ++j) {
            const int gn   = n0 + j * 16 + ncol;
            const float bv = bias[gn];
            const int part = gn >> 10;            // 0=Q 1=K 2=V
            const int wi   = gn & 1023;
            const int head = wi >> 6;
            const int dd   = wi & 63;
#pragma unroll
            for (int r = 0; r < 8; ++r) {
                const int gm = m0 + i * 16 + mb + r;
                const int bi = gm >> 11;          // gm / T_
                const int t  = gm & (T_ - 1);
                const unsigned short val = f2bf(acc[i][j][r] + bv);
                if (part == 0) {
                    Q[(((size_t)bi * HEADS_ + head) * T_ + t) * DK_ + dd] = val;
                } else if (part == 1) {
                    K[(((size_t)bi * HEADS_ + head) * T_ + t) * DK_ + dd] = val;
                } else {
                    // transposed: [B,H,DK,T]
                    Vt[(((size_t)bi * HEADS_ + head) * DK_ + dd) * T_ + t] = val;
                }
            }
        }
    }
}

// ---------------------------------------------------------------------------
// Flash attention (causal). Block = 128 threads = 4 waves; one wave handles
// one head for a 16-row query tile. kv processed in steps of 32.
// ---------------------------------------------------------------------------
__global__ __launch_bounds__(128) void k_attn(const unsigned short* __restrict__ Q,
                                              const unsigned short* __restrict__ K,
                                              const unsigned short* __restrict__ Vt,
                                              unsigned short* __restrict__ ctx) {
    const int lane = threadIdx.x & 31;
    const int wid  = threadIdx.x >> 5;
    const int q0 = blockIdx.x * 16;
    const int h  = blockIdx.y * 4 + wid;
    const int b  = blockIdx.z;

    const unsigned short* Qh = Q  + ((size_t)(b * HEADS_ + h) * T_)  * DK_;
    const unsigned short* Kh = K  + ((size_t)(b * HEADS_ + h) * T_)  * DK_;
    const unsigned short* Vh = Vt + ((size_t)(b * HEADS_ + h) * DK_) * T_;   // [64, T]

    __shared__ unsigned short Plds[4][16 * 32];   // per-wave P staging (C-layout -> A-layout)
    unsigned short* pl = Plds[wid];

    // Q fragments: 16 rows x 64 features = 2 A-frags (k = 0..31, 32..63)
    v16bf aq[2];
#pragma unroll
    for (int ks = 0; ks < 2; ++ks)
        aq[ks] = load_A(Qh + (size_t)q0 * DK_ + ks * 32, DK_, lane);

    const int ncol = lane & 15;
    const int mb   = (lane >> 4) * 8;
    const float scale = 0.125f;                   // 1/sqrt(64)

    float m_i[8], l_i[8];
#pragma unroll
    for (int r = 0; r < 8; ++r) { m_i[r] = -__builtin_inff(); l_i[r] = 0.0f; }
    v8f co[4] = {};                               // context accum over d (4 tiles of 16)

    for (int kv0 = 0; kv0 < q0 + 16; kv0 += 32) {
        // ---- scores: S[16q x 32kv] = Q @ K^T ----
        v8f s0 = {}, s1 = {};
#pragma unroll
        for (int ks = 0; ks < 2; ++ks) {
            v16bf bk0 = load_BT(Kh + (size_t)kv0        * DK_ + ks * 32, DK_, lane);
            v16bf bk1 = load_BT(Kh + (size_t)(kv0 + 16) * DK_ + ks * 32, DK_, lane);
            s0 = wmma_bf16(aq[ks], bk0, s0);
            s1 = wmma_bf16(aq[ks], bk1, s1);
        }
        // ---- scale + causal mask ----
#pragma unroll
        for (int r = 0; r < 8; ++r) {
            const int qrow = q0 + mb + r;
            float v0 = s0[r] * scale; if (kv0 + ncol      > qrow) v0 = -__builtin_inff();
            float v1 = s1[r] * scale; if (kv0 + 16 + ncol > qrow) v1 = -__builtin_inff();
            s0[r] = v0; s1[r] = v1;
        }
        // ---- online softmax: per-row stats via 16-lane xor-shuffle reductions ----
#pragma unroll
        for (int r = 0; r < 8; ++r) {
            float t = fmaxf(s0[r], s1[r]);
#pragma unroll
            for (int o = 8; o >= 1; o >>= 1) t = fmaxf(t, __shfl_xor(t, o, 32));
            const float nm   = fmaxf(m_i[r], t);
            const float corr = __expf(m_i[r] - nm);
            const float p0   = __expf(s0[r] - nm);
            const float p1   = __expf(s1[r] - nm);
            float rs = p0 + p1;
#pragma unroll
            for (int o = 8; o >= 1; o >>= 1) rs += __shfl_xor(rs, o, 32);
            l_i[r] = l_i[r] * corr + rs;
            m_i[r] = nm;
#pragma unroll
            for (int dt = 0; dt < 4; ++dt) co[dt][r] *= corr;
            // stage P (bf16, row-major 16x32) for the PV matmul
            pl[(mb + r) * 32 + ncol]      = f2bf(p0);
            pl[(mb + r) * 32 + 16 + ncol] = f2bf(p1);
        }
        // cross-lane LDS write -> read within the wave: drain DS pipe (CDNA5 wait)
        asm volatile("s_wait_dscnt 0" ::: "memory");
        // ---- context += P @ V : B(k,n) = Vt[(dt*16+n)*T + kv0+k], two 8-half runs per lane ----
        v16bf pa = load_A(pl, 32, lane);
#pragma unroll
        for (int dt = 0; dt < 4; ++dt) {
            v16bf bv = load_BT(Vh + (size_t)(dt * 16) * T_ + kv0, T_, lane);
            co[dt] = wmma_bf16(pa, bv, co[dt]);
        }
    }

    // ---- normalize and store context bf16 at [B,T,DIM] with col = h*64+d ----
#pragma unroll
    for (int r = 0; r < 8; ++r) {
        const float inv = 1.0f / l_i[r];
        const int qrow  = q0 + mb + r;
#pragma unroll
        for (int dt = 0; dt < 4; ++dt)
            ctx[(size_t)(b * T_ + qrow) * DIM_ + h * DK_ + dt * 16 + ncol] =
                f2bf(co[dt][r] * inv);
    }
}

// ---------------------------------------------------------------------------
// Output projection: out = ctx @ w_out^T + b_out (fp32 output)
// ---------------------------------------------------------------------------
__global__ __launch_bounds__(256) void k_out(const unsigned short* __restrict__ cb,
                                             const unsigned short* __restrict__ wb,
                                             const float* __restrict__ bias,
                                             float* __restrict__ out) {
    const int lane = threadIdx.x & 31;
    const int wid  = threadIdx.x >> 5;
    const int m0 = blockIdx.y * 128 + (wid & 3) * 32;
    const int n0 = blockIdx.x * 64  + (wid >> 2) * 32;

    v8f acc[2][2] = {};
    for (int k0 = 0; k0 < DIM_; k0 += 32) {
        v16bf a0 = load_A (cb + (size_t)m0        * DIM_ + k0, DIM_, lane);
        v16bf a1 = load_A (cb + (size_t)(m0 + 16) * DIM_ + k0, DIM_, lane);
        v16bf b0 = load_BT(wb + (size_t)n0        * DIM_ + k0, DIM_, lane);
        v16bf b1 = load_BT(wb + (size_t)(n0 + 16) * DIM_ + k0, DIM_, lane);
        acc[0][0] = wmma_bf16(a0, b0, acc[0][0]);
        acc[0][1] = wmma_bf16(a0, b1, acc[0][1]);
        acc[1][0] = wmma_bf16(a1, b0, acc[1][0]);
        acc[1][1] = wmma_bf16(a1, b1, acc[1][1]);
    }

    const int ncol = lane & 15;
    const int mb   = (lane >> 4) * 8;
#pragma unroll
    for (int i = 0; i < 2; ++i) {
#pragma unroll
        for (int j = 0; j < 2; ++j) {
            const int gn   = n0 + j * 16 + ncol;
            const float bv = bias[gn];
#pragma unroll
            for (int r = 0; r < 8; ++r) {
                const int gm = m0 + i * 16 + mb + r;
                out[(size_t)gm * DIM_ + gn] = acc[i][j][r] + bv;
            }
        }
    }
}

// ---------------------------------------------------------------------------
extern "C" void kernel_launch(void* const* d_in, const int* in_sizes, int n_in,
                              void* d_out, int out_size, void* d_ws, size_t ws_size,
                              hipStream_t stream) {
    (void)in_sizes; (void)n_in; (void)out_size; (void)ws_size;

    const float* x     = (const float*)d_in[0];   // [B,T,DIM]
    const float* w_qkv = (const float*)d_in[1];   // [3D, D]
    const float* b_qkv = (const float*)d_in[2];   // [3D]
    const float* w_out = (const float*)d_in[3];   // [D, D]
    const float* b_out = (const float*)d_in[4];   // [D]
    // d_in[5] = causal mask (tril) — applied analytically in k_attn.

    char* ws = (char*)d_ws;
    size_t off = 0;
    unsigned short* xb    = (unsigned short*)(ws + off); off += (size_t)MROWS_ * DIM_ * 2;
    unsigned short* wqkvb = (unsigned short*)(ws + off); off += (size_t)NQKV_  * DIM_ * 2;
    unsigned short* woutb = (unsigned short*)(ws + off); off += (size_t)DIM_   * DIM_ * 2;
    unsigned short* Qb    = (unsigned short*)(ws + off); off += (size_t)MROWS_ * DIM_ * 2;
    unsigned short* Kb    = (unsigned short*)(ws + off); off += (size_t)MROWS_ * DIM_ * 2;
    unsigned short* Vtb   = (unsigned short*)(ws + off); off += (size_t)MROWS_ * DIM_ * 2;
    unsigned short* ctxb  = (unsigned short*)(ws + off); off += (size_t)MROWS_ * DIM_ * 2;

    // 1) fp32 -> bf16 casts
    k_cast<<<1024, 256, 0, stream>>>(x,     xb,    MROWS_ * DIM_);
    k_cast<<<512,  256, 0, stream>>>(w_qkv, wqkvb, NQKV_  * DIM_);
    k_cast<<<256,  256, 0, stream>>>(w_out, woutb, DIM_   * DIM_);

    // 2) QKV projection (M=8192, N=3072, K=1024)
    k_qkv<<<dim3(NQKV_ / 64, MROWS_ / 128), 256, 0, stream>>>(xb, wqkvb, b_qkv, Qb, Kb, Vtb);

    // 3) causal flash attention
    k_attn<<<dim3(T_ / 16, HEADS_ / 4, B_), 128, 0, stream>>>(Qb, Kb, Vtb, ctxb);

    // 4) output projection (M=8192, N=1024, K=1024), fp32 out
    k_out<<<dim3(DIM_ / 64, MROWS_ / 128), 256, 0, stream>>>(ctxb, woutb, b_out, (float*)d_out);
}
